// CausalGraphBuilder_18219251270369
// MI455X (gfx1250) — compile-verified, matrix operand order checked by reading, not run on hardware
//
#include <hip/hip_runtime.h>
#include <math.h>
#include <stdint.h>

typedef __attribute__((ext_vector_type(16))) _Float16 v16h;
typedef __attribute__((ext_vector_type(8)))  float    v8f;

#define WMMA_F32_F16(a, b, c) \
  __builtin_amdgcn_wmma_f32_16x16x32_f16(false, (a), false, (b), (short)0, (c), false, false)

// A/B fragment (16x32 f16) from LDS f16 matrix, rows contiguous in K.
// Lane L: row = r0 + (L&15); reads K = k0+8*half..+8 and k0+16+8*half..+8 (two b128 DS loads).
__device__ __forceinline__ v16h ldfrag_h(const _Float16* M, int r0, int k0, int ld) {
  int lane = threadIdx.x & 31;
  int row  = r0 + (lane & 15);
  int half = lane >> 4;
  const _Float16* p = M + row * ld + k0 + 8 * half;
  v16h f;
#pragma unroll
  for (int i = 0; i < 8; ++i) { f[i] = p[i]; f[8 + i] = p[16 + i]; }
  return f;
}

// Same fragment but source is f32 in LDS (converted to f16 on load).
__device__ __forceinline__ v16h ldfrag_f(const float* M, int r0, int k0, int ld) {
  int lane = threadIdx.x & 31;
  int row  = r0 + (lane & 15);
  int half = lane >> 4;
  const float* p = M + row * ld + k0 + 8 * half;
  v16h f;
#pragma unroll
  for (int i = 0; i < 8; ++i) { f[i] = (_Float16)p[i]; f[8 + i] = (_Float16)p[16 + i]; }
  return f;
}

// Async global->LDS copy of one 16-byte packet (CDNA5 ASYNCcnt path, bypasses VGPRs).
__device__ __forceinline__ void async_copy_b128(uint32_t lds_addr, const void* gptr) {
  asm volatile("global_load_async_to_lds_b128 %0, %1, off"
               :: "v"(lds_addr), "v"((unsigned long long)(uintptr_t)gptr)
               : "memory");
}
__device__ __forceinline__ void wait_async0() {
  asm volatile("s_wait_asynccnt 0x0" ::: "memory");
}

// ---------------------------------------------------------------------------
// Kernel 1: per-batch node pipeline (one block / batch, 256 thr = 8 waves).
//   nodes(64x256) -> relu(@W1)(64x128) -> @W2+b2(64x64) -> LayerNorm -> nf
//   nf @ [E1a|E1b|S1a|S1b] -> ei,ej (64x64), si,sj (64x32) into workspace
// GEMM1 is double-buffered through registers: next K-chunk's global loads
// are in flight while the current chunk's WMMAs execute.
// ---------------------------------------------------------------------------
__global__ __launch_bounds__(256) void cgb_node_kernel(
    const float* __restrict__ state, const float* __restrict__ W1, const float* __restrict__ b1,
    const float* __restrict__ W2, const float* __restrict__ b2,
    const float* __restrict__ gamma, const float* __restrict__ beta,
    const float* __restrict__ E1, const float* __restrict__ S1,
    float* __restrict__ out_nf,
    float* __restrict__ ei, float* __restrict__ ej,
    float* __restrict__ si, float* __restrict__ sj)
{
  __shared__ __align__(16) _Float16 smA[64 * 64];    //  8 KB: A chunk (nodes)
  __shared__ __align__(16) _Float16 smB[128 * 64];   // 16 KB: transposed B chunks
  __shared__ __align__(16) _Float16 smH1[64 * 128];  // 16 KB: h1 (f16)
  __shared__ __align__(16) float    smH2[64 * 64];   // 16 KB: h2 / node_features (f32)

  const int b    = blockIdx.x;
  const int tid  = threadIdx.x;
  const int wid  = tid >> 5;
  const int lane = tid & 31;
  const v8f z = {0.f, 0.f, 0.f, 0.f, 0.f, 0.f, 0.f, 0.f};

  const int m0  = (wid >> 1) * 16;   // wave's M tile base (0/16/32/48)
  const int nb1 = (wid & 1) * 64;    // wave's N base for 128-wide outputs
  const int nb2 = (wid & 1) * 32;    // wave's N base for  64-wide outputs

  // ================= GEMM1: h1 = relu(nodes @ W1 + b1), K=256 chunked by 64,
  // register double-buffered.
  v8f acc[4];
#pragma unroll
  for (int t = 0; t < 4; ++t) acc[t] = z;

  float regA[16];   // nodes chunk: 4096 elems / 256 thr
  float regB[32];   // W1^T chunk:  8192 elems / 256 thr

  // preload chunk kc=0
#pragma unroll
  for (int u = 0; u < 16; ++u) {
    int e = tid + u * 256;
    int m = e >> 6, kk = e & 63;
    int im = (m * 127) / 63;                          // linspace(0,127,64) floor
    regA[u] = state[((size_t)b * 128 + im) * 256 + kk];
  }
#pragma unroll
  for (int u = 0; u < 32; ++u) {
    int e = tid + u * 256;
    int kk = e >> 7, n = e & 127;                     // n-major per lane: coalesced
    regB[u] = W1[(size_t)kk * 128 + n];
  }

  for (int kc = 0; kc < 256; kc += 64) {
    // commit staged regs to LDS (f16)
#pragma unroll
    for (int u = 0; u < 16; ++u) smA[tid + u * 256] = (_Float16)regA[u];
#pragma unroll
    for (int u = 0; u < 32; ++u) {
      int e = tid + u * 256;
      int kk = e >> 7, n = e & 127;
      smB[n * 64 + kk] = (_Float16)regB[u];
    }
    // launch next chunk's global loads; they overlap the WMMAs below
    if (kc < 192) {
      int kn = kc + 64;
#pragma unroll
      for (int u = 0; u < 16; ++u) {
        int e = tid + u * 256;
        int m = e >> 6, kk = e & 63;
        int im = (m * 127) / 63;
        regA[u] = state[((size_t)b * 128 + im) * 256 + kn + kk];
      }
#pragma unroll
      for (int u = 0; u < 32; ++u) {
        int e = tid + u * 256;
        int kk = e >> 7, n = e & 127;
        regB[u] = W1[(size_t)(kn + kk) * 128 + n];
      }
    }
    __syncthreads();
#pragma unroll
    for (int ks = 0; ks < 64; ks += 32) {
      v16h a = ldfrag_h(smA, m0, ks, 64);
#pragma unroll
      for (int t = 0; t < 4; ++t) {
        v16h bb = ldfrag_h(smB, nb1 + 16 * t, ks, 64);
        acc[t] = WMMA_F32_F16(a, bb, acc[t]);
      }
    }
    __syncthreads();
  }
  {
    int mb = m0 + ((lane >> 4) << 3);
#pragma unroll
    for (int t = 0; t < 4; ++t) {
      int n = nb1 + 16 * t + (lane & 15);
      float bias = b1[n];
#pragma unroll
      for (int r = 0; r < 8; ++r)
        smH1[(mb + r) * 128 + n] = (_Float16)fmaxf(acc[t][r] + bias, 0.f);
    }
  }
  __syncthreads();

  // ================= GEMM2: h2 = h1 @ W2 + b2  (64x128x64)
  for (int e = tid; e < 64 * 128; e += 256) {           // W2^T: [n][k], coalesced reads
    int k = e >> 6, n = e & 63;
    smB[n * 128 + k] = (_Float16)W2[(size_t)k * 64 + n];
  }
  __syncthreads();
  v8f acc2[2]; acc2[0] = z; acc2[1] = z;
#pragma unroll
  for (int ks = 0; ks < 128; ks += 32) {
    v16h a = ldfrag_h(smH1, m0, ks, 128);
#pragma unroll
    for (int t = 0; t < 2; ++t) {
      v16h bb = ldfrag_h(smB, nb2 + 16 * t, ks, 128);
      acc2[t] = WMMA_F32_F16(a, bb, acc2[t]);
    }
  }
  {
    int mb = m0 + ((lane >> 4) << 3);
#pragma unroll
    for (int t = 0; t < 2; ++t) {
      int n = nb2 + 16 * t + (lane & 15);
      float bias = b2[n];
#pragma unroll
      for (int r = 0; r < 8; ++r)
        smH2[(mb + r) * 64 + n] = acc2[t][r] + bias;
    }
  }
  __syncthreads();

  // ================= LayerNorm(64) + nan_to_num -> node_features (f32, in smH2)
  if (tid < 64) {
    float mu = 0.f;
#pragma unroll
    for (int d = 0; d < 64; ++d) mu += smH2[tid * 64 + d];
    mu *= (1.f / 64.f);
    float var = 0.f;
#pragma unroll
    for (int d = 0; d < 64; ++d) { float x = smH2[tid * 64 + d] - mu; var += x * x; }
    var *= (1.f / 64.f);
    float inv = rsqrtf(var + 1e-5f);
    for (int d = 0; d < 64; ++d) {
      float v = (smH2[tid * 64 + d] - mu) * inv * gamma[d] + beta[d];
      if (v != v) v = 0.f;  // nan_to_num(nan=0)
      smH2[tid * 64 + d] = v;
      out_nf[((size_t)b * 64 + tid) * 64 + d] = v;
    }
  }
  __syncthreads();

  // ================= GEMM3 pass 0: nf @ [E1a | E1b] -> ei, ej (64x64 each)
  for (int e = tid; e < 128 * 64; e += 256) {           // PT0: [n'][k], coalesced reads
    int k = e >> 7, np = e & 127;
    float v = (np < 64) ? E1[(size_t)k * 64 + np] : E1[(size_t)(64 + k) * 64 + (np - 64)];
    smB[np * 64 + k] = (_Float16)v;
  }
  __syncthreads();
  v8f acc3[4];
#pragma unroll
  for (int t = 0; t < 4; ++t) acc3[t] = z;
#pragma unroll
  for (int ks = 0; ks < 64; ks += 32) {
    v16h a = ldfrag_f(smH2, m0, ks, 64);
#pragma unroll
    for (int t = 0; t < 4; ++t) {
      v16h bb = ldfrag_h(smB, nb1 + 16 * t, ks, 64);
      acc3[t] = WMMA_F32_F16(a, bb, acc3[t]);
    }
  }
  {
    int mb = m0 + ((lane >> 4) << 3);
#pragma unroll
    for (int t = 0; t < 4; ++t) {
      int np = nb1 + 16 * t + (lane & 15);
#pragma unroll
      for (int r = 0; r < 8; ++r) {
        int m = mb + r;
        float v = acc3[t][r];
        if (np < 64) ei[((size_t)b * 64 + m) * 64 + np] = v;
        else         ej[((size_t)b * 64 + m) * 64 + (np - 64)] = v;
      }
    }
  }
  __syncthreads();

  // ================= GEMM3 pass 1: nf @ [S1a | S1b] -> si, sj (64x32 each)
  for (int e = tid; e < 64 * 64; e += 256) {            // PT1: [n'][k], coalesced reads
    int k = e >> 6, np = e & 63;
    float v = (np < 32) ? S1[(size_t)k * 32 + np] : S1[(size_t)(64 + k) * 32 + (np - 32)];
    smB[np * 64 + k] = (_Float16)v;
  }
  __syncthreads();
  v8f acc4[2]; acc4[0] = z; acc4[1] = z;
#pragma unroll
  for (int ks = 0; ks < 64; ks += 32) {
    v16h a = ldfrag_f(smH2, m0, ks, 64);
#pragma unroll
    for (int t = 0; t < 2; ++t) {
      v16h bb = ldfrag_h(smB, nb2 + 16 * t, ks, 64);
      acc4[t] = WMMA_F32_F16(a, bb, acc4[t]);
    }
  }
  {
    int mb = m0 + ((lane >> 4) << 3);
#pragma unroll
    for (int t = 0; t < 2; ++t) {
      int np = nb2 + 16 * t + (lane & 15);
#pragma unroll
      for (int r = 0; r < 8; ++r) {
        int m = mb + r;
        float v = acc4[t][r];
        if (np < 32) si[((size_t)b * 64 + m) * 32 + np] = v;
        else         sj[((size_t)b * 64 + m) * 32 + (np - 32)] = v;
      }
    }
  }
}

// ---------------------------------------------------------------------------
// Kernel 2: fused pairwise edge/strength MLP. One block per (i, b):
//   ej/sj tiles staged global->LDS with GLOBAL_LOAD_ASYNC_TO_LDS_B128
//   A = relu(ei[b,i] + ej[b,:] + bE1)  (64 j x 64 h, f16 in LDS)
//   he2 = relu(A @ E2 + bE2) via WMMA  (64 x 32, f32)
//   adjacency[j] = thresh(sigmoid(he2[j]·E3 + bE3) * offdiag)
//   strength[j]  = tanh(relu(si[i]+sj[j]+bS1)·S2 + bS2) * offdiag
// 128 threads = 4 waves; wave w owns j-rows [16w, 16w+16).
// ---------------------------------------------------------------------------
__global__ __launch_bounds__(128) void cgb_edge_kernel(
    const float* __restrict__ ei, const float* __restrict__ ej,
    const float* __restrict__ si, const float* __restrict__ sj,
    const float* __restrict__ bE1, const float* __restrict__ E2, const float* __restrict__ bE2,
    const float* __restrict__ E3, const float* __restrict__ bE3,
    const float* __restrict__ bS1, const float* __restrict__ S2, const float* __restrict__ bS2,
    float* __restrict__ out_adj, float* __restrict__ out_str)
{
  __shared__ __align__(16) float    smEj[64 * 64];   // 16 KB, filled by async DMA
  __shared__ __align__(16) float    smSj[64 * 32];   //  8 KB, filled by async DMA
  __shared__ __align__(16) _Float16 smHe[64 * 64];   //  8 KB A matrix
  __shared__ __align__(16) _Float16 smE2T[32 * 64];  //  4 KB E2^T [n][k]
  __shared__ __align__(16) float    smHe2[64 * 32];  //  8 KB hidden-2
  __shared__ float smEi[64], smBE1v[64], smSi[32], smBS1[32], smS2[32], smE3[32];

  const int i    = blockIdx.x;
  const int b    = blockIdx.y;
  const int tid  = threadIdx.x;
  const int wid  = tid >> 5;
  const int lane = tid & 31;
  const v8f z = {0.f, 0.f, 0.f, 0.f, 0.f, 0.f, 0.f, 0.f};

  // ---- 1) kick off async global->LDS DMA of ej (16 KB) and sj (8 KB) tiles
  {
    const uint32_t lEj = (uint32_t)(uintptr_t)&smEj[0];
    const uint32_t lSj = (uint32_t)(uintptr_t)&smSj[0];
    const float* gEj = ej + (size_t)b * 64 * 64;
    const float* gSj = sj + (size_t)b * 64 * 32;
#pragma unroll
    for (int u = 0; u < 8; ++u) {                 // 1024 x 16B packets / 128 thr
      int p = tid + u * 128;
      async_copy_b128(lEj + p * 16, gEj + p * 4);
    }
#pragma unroll
    for (int u = 0; u < 4; ++u) {                 // 512 x 16B packets / 128 thr
      int p = tid + u * 128;
      async_copy_b128(lSj + p * 16, gSj + p * 4);
    }
  }

  // ---- 2) regular small staging while the DMA is in flight
  if (tid < 64) { smEi[tid] = ei[((size_t)b * 64 + i) * 64 + tid]; smBE1v[tid] = bE1[tid]; }
  else {
    int k = tid - 64;
    if (k < 32) { smSi[k] = si[((size_t)b * 64 + i) * 32 + k]; smBS1[k] = bS1[k]; }
    else        { int k2 = k - 32; smS2[k2] = S2[k2]; smE3[k2] = E3[k2]; }
  }
  for (int e = tid; e < 32 * 64; e += 128) {      // E2^T [n][k], coalesced reads
    int k = e >> 5, n = e & 31;
    smE2T[n * 64 + k] = (_Float16)E2[(size_t)k * 32 + n];
  }

  // ---- 3) each wave drains its own async writes, then barrier
  wait_async0();
  __syncthreads();

  // ---- 4) A = relu(ei_i + ej_j + bE1), f16
  for (int e = tid; e < 64 * 64; e += 128) {
    int h = e & 63;
    float v = smEi[h] + smEj[e] + smBE1v[h];
    smHe[e] = (_Float16)fmaxf(v, 0.f);
  }
  __syncthreads();

  // ---- 5) he2 = relu(A @ E2 + bE2) via WMMA
  const int m0 = wid * 16;
  v8f acc[2]; acc[0] = z; acc[1] = z;
#pragma unroll
  for (int ks = 0; ks < 64; ks += 32) {
    v16h a = ldfrag_h(smHe, m0, ks, 64);
#pragma unroll
    for (int t = 0; t < 2; ++t) {
      v16h bb = ldfrag_h(smE2T, 16 * t, ks, 64);
      acc[t] = WMMA_F32_F16(a, bb, acc[t]);
    }
  }
  {
    int mb = m0 + ((lane >> 4) << 3);
#pragma unroll
    for (int t = 0; t < 2; ++t) {
      int n = 16 * t + (lane & 15);
      float bias = bE2[n];
#pragma unroll
      for (int r = 0; r < 8; ++r)
        smHe2[(mb + r) * 32 + n] = fmaxf(acc[t][r] + bias, 0.f);
    }
  }
  __syncthreads();

  // ---- 6) heads
  if (tid < 64) {                               // edge head
    int j = tid;
    float s = bE3[0];
#pragma unroll
    for (int k = 0; k < 32; ++k) s += smHe2[j * 32 + k] * smE3[k];
    float p = 1.f / (1.f + expf(-s));
    if (j == i) p = 0.f;
    out_adj[((size_t)b * 64 + i) * 64 + j] = (p > 0.3f) ? p : 0.f;
  } else {                                      // strength head
    int j = tid - 64;
    float ss = bS2[0];
#pragma unroll
    for (int k = 0; k < 32; ++k) {
      float v = smSi[k] + smSj[j * 32 + k] + smBS1[k];
      ss += fmaxf(v, 0.f) * smS2[k];
    }
    float st = tanhf(ss);
    if (j == i) st = 0.f;
    out_str[((size_t)b * 64 + i) * 64 + j] = st;
  }
}

extern "C" void kernel_launch(void* const* d_in, const int* in_sizes, int n_in,
                              void* d_out, int out_size, void* d_ws, size_t ws_size,
                              hipStream_t stream) {
  (void)in_sizes; (void)n_in; (void)out_size; (void)ws_size;
  const float* state = (const float*)d_in[0];
  const float* W1    = (const float*)d_in[1];
  const float* b1    = (const float*)d_in[2];
  const float* W2    = (const float*)d_in[3];
  const float* b2    = (const float*)d_in[4];
  const float* gamma = (const float*)d_in[5];
  const float* beta  = (const float*)d_in[6];
  const float* E1    = (const float*)d_in[7];
  const float* bE1   = (const float*)d_in[8];
  const float* E2    = (const float*)d_in[9];
  const float* bE2   = (const float*)d_in[10];
  const float* E3    = (const float*)d_in[11];
  const float* bE3   = (const float*)d_in[12];
  const float* S1    = (const float*)d_in[13];
  const float* bS1   = (const float*)d_in[14];
  const float* S2    = (const float*)d_in[15];
  const float* bS2   = (const float*)d_in[16];

  float* out_nf  = (float*)d_out;                      // (256,64,64)
  float* out_adj = out_nf + (size_t)256 * 64 * 64;     // (256,64,64)
  float* out_str = out_adj + (size_t)256 * 64 * 64;    // (256,64,64)

  float* ei = (float*)d_ws;                            // (256,64,64)
  float* ej = ei + (size_t)256 * 64 * 64;              // (256,64,64)
  float* si = ej + (size_t)256 * 64 * 64;              // (256,64,32)
  float* sj = si + (size_t)256 * 64 * 32;              // (256,64,32)

  cgb_node_kernel<<<256, 256, 0, stream>>>(state, W1, b1, W2, b2, gamma, beta, E1, S1,
                                           out_nf, ei, ej, si, sj);
  cgb_edge_kernel<<<dim3(64, 256), 128, 0, stream>>>(ei, ej, si, sj, bE1, E2, bE2, E3, bE3,
                                                     bS1, S2, bS2, out_adj, out_str);
}
